// ChamferDistance_58308476010760
// MI455X (gfx1250) — matrix operand order repaired and checked by reading, not compiled, not run on hardware
//
#include <hip/hip_runtime.h>
#include <hip/hip_bf16.h>

// ---------------------------------------------------------------------------
// Chamfer distance, CDNA5 (gfx1250), bf16x2-split WMMA GEMM + fused min-reduce
// ---------------------------------------------------------------------------

typedef __attribute__((ext_vector_type(8)))  float  v8f;
typedef __attribute__((ext_vector_type(16))) __bf16 v16bf;
typedef __attribute__((ext_vector_type(8)))  __bf16 v8bf;

union V16 { v16bf v; v8bf h[2]; };

#define BATCH  64
#define NV     2048
#define NL     512
#define KDIM   512
#define TV     64          // video rows per workgroup
#define TL     64          // lang rows per LDS chunk
#define AK     520         // padded K stride (elems): 1040B row pitch -> 4-bank step
#define NCHUNK (NL / TL)   // 8
#define KC     (KDIM / 32) // 16 K-chunks of 32

// LDS layout (bytes):
//   Ahi[TV*AK] Alo[TV*AK] Bhi[TL*AK] Blo[TL*AK]  (u16 planes) = 266240
//   a2s[64] b2s[64] colPart[256] rowHalf[128] red[1] (f32)    = 2052
#define LDS_BYTES 268304

#define WMMA_BF16(A, B, C) \
  __builtin_amdgcn_wmma_f32_16x16x32_bf16(false, (A), false, (B), (short)0, (C), false, false)

__device__ __forceinline__ void split_bf16x2(float v, unsigned& hb, unsigned& lb) {
  unsigned u = __float_as_uint(v);
  hb = (u + 0x7FFFu + ((u >> 16) & 1u)) >> 16;           // RNE bf16 of v
  float r = v - __uint_as_float(hb << 16);               // residual
  unsigned ur = __float_as_uint(r);
  lb = (ur + 0x7FFFu + ((ur >> 16) & 1u)) >> 16;         // RNE bf16 of residual
}

// Stage a 64x512 f32 block from global into hi/lo bf16 LDS planes (stride AK),
// accumulating per-row sum-of-squares into sq[] via ds_add_f32.
__device__ __forceinline__ void stage64x512(const float* __restrict__ gsrc,
                                            unsigned short* hiP, unsigned short* loP,
                                            float* sq, int t) {
  int row  = t >> 2;             // 64 rows, 4 threads each
  int cseg = (t & 3) * 128;      // 128 contiguous K per thread
  const float* gp = gsrc + (size_t)row * KDIM + cseg;
  unsigned short* hp = hiP + row * AK + cseg;
  unsigned short* lp = loP + row * AK + cseg;
  float ss = 0.f;
#pragma unroll 4
  for (int i = 0; i < 128; i += 4) {
    float4 x = *(const float4*)(gp + i);
    ss += x.x * x.x + x.y * x.y + x.z * x.z + x.w * x.w;
    unsigned h0, h1, h2, h3, l0, l1, l2, l3;
    split_bf16x2(x.x, h0, l0); split_bf16x2(x.y, h1, l1);
    split_bf16x2(x.z, h2, l2); split_bf16x2(x.w, h3, l3);
    *(uint2*)(hp + i) = make_uint2(h0 | (h1 << 16), h2 | (h3 << 16));
    *(uint2*)(lp + i) = make_uint2(l0 | (l1 << 16), l2 | (l3 << 16));
  }
  atomicAdd(&sq[row], ss);
}

__global__ void chamfer_init(float* gcol, float* rowSum) {
  int i = blockIdx.x * 256 + threadIdx.x;
  if (i < BATCH * NL) gcol[i] = __builtin_inff();
  if (i < BATCH)      rowSum[i] = 0.f;
}

__global__ __launch_bounds__(256, 1)
void chamfer_main(const float* __restrict__ video, const float* __restrict__ lang,
                  float* gcol, float* rowSum) {
  extern __shared__ char smem[];
  unsigned short* Ahi = (unsigned short*)smem;
  unsigned short* Alo = Ahi + TV * AK;
  unsigned short* Bhi = Alo + TV * AK;
  unsigned short* Blo = Bhi + TL * AK;
  float* a2s     = (float*)(Blo + TL * AK);  // 64
  float* b2s     = a2s + TV;                 // 64
  float* colPart = b2s + TL;                 // 4*64
  float* rowHalf = colPart + 4 * TL;         // 2*64
  float* redp    = rowHalf + 2 * TV;         // 1

  const int b     = blockIdx.x >> 5;    // 32 v-blocks per batch
  const int vb    = blockIdx.x & 31;
  const int vbase = vb * TV;
  const int t     = threadIdx.x;
  const int lane  = t & 31;
  const int wave  = t >> 5;
  const int mt    = wave & 3;   // m-tile (16 rows) owned by wave
  const int ntg   = wave >> 2;  // n-tile group: handles nt = 2*ntg + {0,1}
  const int l15   = lane & 15;
  const int half  = lane >> 4;

  if (t < TV) a2s[t] = 0.f;
  if (t == 0) *redp = 0.f;
  __syncthreads();

  // Stage the workgroup's video block once (full K resident).
  stage64x512(video + ((size_t)b * NV + vbase) * KDIM, Ahi, Alo, a2s, t);
  __syncthreads();

  // Per-wave persistent state.
  float rowAcc[8];
#pragma unroll
  for (int i = 0; i < 8; ++i) rowAcc[i] = __builtin_inff();
  float a2v[8];
#pragma unroll
  for (int i = 0; i < 8; ++i) a2v[i] = a2s[mt * 16 + half * 8 + i];  // C-row layout

  const int arowOff  = (mt * 16 + l15) * AK + 16 * half;    // 16-bit A frag: K split 16/16
  const int browOff0 = (ntg * 32 + l15) * AK + 16 * half;   // 16-bit B frag: K split 16/16
  const int browOff1 = browOff0 + 16 * AK;

  for (int ch = 0; ch < NCHUNK; ++ch) {
    const int nbase = ch * TL;
    if (t < TL) b2s[t] = 0.f;
    __syncthreads();
    stage64x512(lang + ((size_t)b * NL + nbase) * KDIM, Bhi, Blo, b2s, t);
    __syncthreads();

    v8f acc[2];
    acc[0] = (v8f){0.f, 0.f, 0.f, 0.f, 0.f, 0.f, 0.f, 0.f};
    acc[1] = acc[0];

#pragma unroll
    for (int kc = 0; kc < KC; ++kc) {
      const int ao = arowOff + kc * 32;
      V16 Ah, Al;
      Ah.h[0] = *(const v8bf*)(Ahi + ao);
      Ah.h[1] = *(const v8bf*)(Ahi + ao + 8);
      Al.h[0] = *(const v8bf*)(Alo + ao);
      Al.h[1] = *(const v8bf*)(Alo + ao + 8);
#pragma unroll
      for (int n = 0; n < 2; ++n) {
        const int bo = (n ? browOff1 : browOff0) + kc * 32;
        V16 Bh, Bl;
        Bh.h[0] = *(const v8bf*)(Bhi + bo);
        Bh.h[1] = *(const v8bf*)(Bhi + bo + 8);
        Bl.h[0] = *(const v8bf*)(Blo + bo);
        Bl.h[1] = *(const v8bf*)(Blo + bo + 8);
        acc[n] = WMMA_BF16(Ah.v, Bh.v, acc[n]);   // hi*hi
        acc[n] = WMMA_BF16(Ah.v, Bl.v, acc[n]);   // hi*lo
        acc[n] = WMMA_BF16(Al.v, Bh.v, acc[n]);   // lo*hi
      }
    }

    // pairwise = a2 + b2 - 2ab; fold a2/b2 separately for the two reductions.
#pragma unroll
    for (int n = 0; n < 2; ++n) {
      const int nt = ntg * 2 + n;
      const float b2c = b2s[nt * 16 + l15];
      float colv = __builtin_inff();
#pragma unroll
      for (int i = 0; i < 8; ++i) {
        const float m2c = -2.0f * acc[n][i];
        rowAcc[i] = fminf(rowAcc[i], b2c + m2c);   // min over l of (b2 - 2ab)
        colv      = fminf(colv, a2v[i] + m2c);     // min over v of (a2 - 2ab)
      }
      colv = fminf(colv, __shfl_xor(colv, 16, 32));  // combine M halves
      if (lane < 16) colPart[mt * TL + nt * 16 + lane] = colv;
    }
    __syncthreads();

    if (t < TL) {  // combine the 4 m-tiles, then device-scope float atomic min
      float m = fminf(fminf(colPart[t], colPart[TL + t]),
                      fminf(colPart[2 * TL + t], colPart[3 * TL + t]));
      float* dst = gcol + (size_t)b * NL + nbase + t;
      asm volatile("global_atomic_min_num_f32 %0, %1, off scope:SCOPE_DEV"
                   :: "v"(dst), "v"(m) : "memory");
    }
    __syncthreads();
  }

  // Finalize row mins: butterfly across the 16 lanes of each half.
#pragma unroll
  for (int i = 0; i < 8; ++i) {
    float v = rowAcc[i];
    v = fminf(v, __shfl_xor(v, 1, 32));
    v = fminf(v, __shfl_xor(v, 2, 32));
    v = fminf(v, __shfl_xor(v, 4, 32));
    v = fminf(v, __shfl_xor(v, 8, 32));
    rowAcc[i] = v;
  }
  if (l15 == 0) {
#pragma unroll
    for (int i = 0; i < 8; ++i)
      rowHalf[ntg * TV + mt * 16 + half * 8 + i] = rowAcc[i];
  }
  __syncthreads();
  if (t < TV) {
    float v = fminf(rowHalf[t], rowHalf[TV + t]) + a2s[t];  // + ||video_v||^2
    atomicAdd(redp, v);
  }
  __syncthreads();
  if (t == 0) atomicAdd(&rowSum[b], *redp);
}

__global__ __launch_bounds__(256)
void chamfer_final(const float* __restrict__ lang, const float* __restrict__ gcol,
                   const float* __restrict__ rowSum, float* __restrict__ out) {
  __shared__ float red;
  const int b = blockIdx.x, t = threadIdx.x;
  if (t == 0) red = 0.f;
  __syncthreads();
  float part = 0.f;
  for (int l = t; l < NL; l += 256) {
    const float* lp = lang + ((size_t)b * NL + l) * KDIM;
    float ss = 0.f;
#pragma unroll 4
    for (int i = 0; i < KDIM; i += 4) {
      float4 x = *(const float4*)(lp + i);
      ss += x.x * x.x + x.y * x.y + x.z * x.z + x.w * x.w;
    }
    part += gcol[(size_t)b * NL + l] + ss;   // min_v(a2-2ab) + ||lang_l||^2
  }
  atomicAdd(&red, part);
  __syncthreads();
  if (t == 0) out[b] = rowSum[b] * (1.0f / NV) + red * (1.0f / NL);
}

extern "C" void kernel_launch(void* const* d_in, const int* in_sizes, int n_in,
                              void* d_out, int out_size, void* d_ws, size_t ws_size,
                              hipStream_t stream) {
  (void)in_sizes; (void)n_in; (void)out_size; (void)ws_size;
  const float* video = (const float*)d_in[0];
  const float* lang  = (const float*)d_in[1];
  float* out    = (float*)d_out;
  float* gcol   = (float*)d_ws;          // [64][512] column mins (min_v of a2-2ab)
  float* rowSum = gcol + BATCH * NL;     // [64]      sum_v of row mins

  chamfer_init<<<(BATCH * NL + 255) / 256, 256, 0, stream>>>(gcol, rowSum);
  chamfer_main<<<dim3(BATCH * 32), 256, LDS_BYTES, stream>>>(video, lang, gcol, rowSum);
  chamfer_final<<<BATCH, 256, 0, stream>>>(lang, gcol, rowSum, out);
}
